// NormalRefinementTransformer_17265768530258
// MI455X (gfx1250) — compile-verified, matrix operand order checked by reading, not run on hardware
//
#include <hip/hip_runtime.h>
#include <math.h>

// ---------------- problem constants (match reference) ----------------
#define BATCH 2
#define NPTS  8192
#define CH    128
#define NL    3
#define NK    16
#define BN    (BATCH * NPTS)   // 16384 total points

typedef _Float16 hv16 __attribute__((ext_vector_type(16)));
typedef _Float16 hv8  __attribute__((ext_vector_type(8)));
typedef float    fv8  __attribute__((ext_vector_type(8)));

// ---------------------------------------------------------------------
// WMMA fragment layouts (CDNA5 16x16x32 f16, CDNA5 ISA)
//   A (16x32 f16): lane<16 -> M=lane,    K = e<8 ? e   : e+8
//                  lane>=16 -> M=lane-16, K = e<8 ? e+8 : e+16
//   B (32x16 f16): same K pattern, N = lane&15
//   C/D (16x16 f32): lane l, reg r -> M = r + 8*(l>=16), N = l&15
// ---------------------------------------------------------------------
__device__ __forceinline__ int kmap(int e, int khi) {
  return (e < 8) ? (e + 8 * khi) : (e + 8 + 8 * khi);
}

__device__ __forceinline__ fv8 wmma_f16(hv16 a, hv16 b, fv8 c) {
  return __builtin_amdgcn_wmma_f32_16x16x32_f16(false, a, false, b, (short)0, c,
                                                false, false);
}

// Fast A loader: row-major f32 [M x lda], K-range fully valid (padded).
// 16 elements per lane = 4x float4 contiguous loads, no predication.
__device__ __forceinline__ hv16 load_a_fast(const float* __restrict__ A, int lda,
                                            int m0, int k0, int lane) {
  int m = m0 + (lane & 15);
  int khi = (lane >> 4) & 1;
  const float* p = A + (long long)m * lda + k0 + 8 * khi;
  float4 a0 = *(const float4*)(p + 0);
  float4 a1 = *(const float4*)(p + 4);
  float4 a2 = *(const float4*)(p + 16);
  float4 a3 = *(const float4*)(p + 20);
  hv16 r;
  r[0] = (_Float16)a0.x;  r[1] = (_Float16)a0.y;
  r[2] = (_Float16)a0.z;  r[3] = (_Float16)a0.w;
  r[4] = (_Float16)a1.x;  r[5] = (_Float16)a1.y;
  r[6] = (_Float16)a1.z;  r[7] = (_Float16)a1.w;
  r[8] = (_Float16)a2.x;  r[9] = (_Float16)a2.y;
  r[10] = (_Float16)a2.z; r[11] = (_Float16)a2.w;
  r[12] = (_Float16)a3.x; r[13] = (_Float16)a3.y;
  r[14] = (_Float16)a3.z; r[15] = (_Float16)a3.w;
  return r;
}

// Packed-B fragment load: one contiguous 32B load per lane (coalesced 1KB/wave)
__device__ __forceinline__ hv16 load_b_packed(const _Float16* __restrict__ Wp,
                                              int tn, int kc, int nt, int lane) {
  return *(const hv16*)(Wp + (((long long)kc * tn + nt) * 32 + lane) * 16);
}

// A fragment from per-wave f16 LDS buffer [16 x CH]: two 16B LDS loads
__device__ __forceinline__ hv16 load_a_lds(const _Float16* buf, int k0, int lane) {
  int m = lane & 15;
  int khi = (lane >> 4) & 1;
  const _Float16* p = buf + m * CH + k0 + 8 * khi;
  hv8 lo = *(const hv8*)(p);
  hv8 hi = *(const hv8*)(p + 16);
  hv16 r;
#pragma unroll
  for (int e = 0; e < 8; ++e) { r[e] = lo[e]; r[8 + e] = hi[e]; }
  return r;
}

// ---------------------------------------------------------------------
// One-time weight pack: row-major f32 W[K x N] -> f16 fragments
// Wp[kc][nt][lane][16] matching the B layout, zero-padded in K and N.
// ---------------------------------------------------------------------
__global__ void pack_w_kernel(const float* __restrict__ W, int K, int N,
                              _Float16* __restrict__ Wp) {
  int kcN = (K + 31) >> 5;
  int tn = (N + 15) >> 4;
  int total = kcN * tn * 32;
  int t = blockIdx.x * blockDim.x + threadIdx.x;
  if (t >= total) return;
  int lane = t & 31;
  int tile = t >> 5;
  int nt = tile % tn;
  int kc = tile / tn;
  int n = (nt << 4) + (lane & 15);
  int khi = (lane >> 4) & 1;
  _Float16* dst = Wp + (long long)t * 16;
#pragma unroll
  for (int e = 0; e < 16; ++e) {
    int kk = (kc << 5) + kmap(e, khi);
    float v = (n < N && kk < K) ? W[(long long)kk * N + n] : 0.0f;
    dst[e] = (_Float16)v;
  }
}

// ---------------------------------------------------------------------
// Generic WMMA linear: Out[M x N] = act(A[M x K] @ W + bias) (+ residual)
// A padded so that kcN*32 columns are readable. act: 0=none 1=relu 2=sigmoid.
// One wave computes one 16x16 tile; M must be a multiple of 16.
// ---------------------------------------------------------------------
__global__ void linear_wmma_kernel(const float* __restrict__ A, int lda,
                                   const _Float16* __restrict__ Wp,
                                   const float* __restrict__ bias,
                                   const float* __restrict__ residual,
                                   float* __restrict__ Out,
                                   int M, int kcN, int N, int act) {
  const int lane = threadIdx.x & 31;
  const int wave = threadIdx.x >> 5;
  const int wpb = blockDim.x >> 5;
  const int tn = (N + 15) >> 4;
  const int tm = M >> 4;
  const long long total = (long long)tn * tm;
  const long long stride = (long long)gridDim.x * wpb;

  for (long long t = (long long)blockIdx.x * wpb + wave; t < total; t += stride) {
    int nt = (int)(t % tn);
    int mt = (int)(t / tn);
    int m0 = mt << 4, n0 = nt << 4;
    fv8 acc = {};
    for (int kc = 0; kc < kcN; ++kc) {
      hv16 a = load_a_fast(A, lda, m0, kc << 5, lane);
      hv16 b = load_b_packed(Wp, tn, kc, nt, lane);
      acc = wmma_f16(a, b, acc);
    }
    int n = n0 + (lane & 15);
    int khi = (lane >> 4) & 1;
    if (n < N) {
      float bv = bias ? bias[n] : 0.0f;
#pragma unroll
      for (int r = 0; r < 8; ++r) {
        int m = m0 + r + 8 * khi;
        float v = acc[r] + bv;
        if (act == 1) v = fmaxf(v, 0.0f);
        else if (act == 2) v = 1.0f / (1.0f + __expf(-v));
        if (residual) v += residual[(long long)m * N + n];
        Out[(long long)m * N + n] = v;
      }
    }
  }
}

// ---------------------------------------------------------------------
// KNN (top-17 incl self) + curvature (first 8 neighbors).
// One thread per query point, 256 queries/block, candidates tiled via LDS.
// idxOut holds GLOBAL point ids (b*NPTS + j).
// ---------------------------------------------------------------------
__global__ void knn_curv_kernel(const float* __restrict__ xyz,
                                const float* __restrict__ normals,
                                int* __restrict__ idxOut,
                                float* __restrict__ curvOut) {
  __shared__ float sx[256], sy[256], sz[256];
  const int b = blockIdx.y;
  const int q = blockIdx.x * 256 + threadIdx.x;
  const int g = b * NPTS + q;

  const float qx = xyz[(long long)g * 3 + 0];
  const float qy = xyz[(long long)g * 3 + 1];
  const float qz = xyz[(long long)g * 3 + 2];

  float bestd[17];
  int   besti[17];
#pragma unroll
  for (int i = 0; i < 17; ++i) { bestd[i] = 3.0e38f; besti[i] = g; }

  for (int t0 = 0; t0 < NPTS; t0 += 256) {
    int cg = b * NPTS + t0 + threadIdx.x;
    sx[threadIdx.x] = xyz[(long long)cg * 3 + 0];
    sy[threadIdx.x] = xyz[(long long)cg * 3 + 1];
    sz[threadIdx.x] = xyz[(long long)cg * 3 + 2];
    __syncthreads();
    if (t0 + 256 < NPTS)
      __builtin_prefetch(xyz + (long long)(b * NPTS + t0 + 256 + threadIdx.x) * 3, 0, 0);
    for (int j = 0; j < 256; ++j) {
      float dx = qx - sx[j], dy = qy - sy[j], dz = qz - sz[j];
      float d = dx * dx + dy * dy + dz * dz;
      if (d < bestd[16]) {
        int p = 16;
        while (p > 0 && bestd[p - 1] > d) {
          bestd[p] = bestd[p - 1];
          besti[p] = besti[p - 1];
          --p;
        }
        bestd[p] = d;
        besti[p] = b * NPTS + t0 + j;
      }
    }
    __syncthreads();
  }

  for (int j = 0; j < NK; ++j) idxOut[(long long)g * NK + j] = besti[j + 1];

  const float nx = normals[(long long)g * 3 + 0];
  const float ny = normals[(long long)g * 3 + 1];
  const float nz = normals[(long long)g * 3 + 2];
  float s = 0.0f;
  for (int j = 1; j <= 8; ++j) {
    long long h = besti[j];
    float cx = normals[h * 3 + 0], cy = normals[h * 3 + 1], cz = normals[h * 3 + 2];
    s += fabsf(nx * cx + ny * cy + nz * cz);
  }
  curvOut[g] = 1.0f - s * 0.125f;
}

// pack [xyz, normals, curvature] -> (BN x 32), columns 7..31 zeroed
__global__ void pack_in7_kernel(const float* __restrict__ xyz,
                                const float* __restrict__ normals,
                                const float* __restrict__ curv,
                                float* __restrict__ in7) {
  int g = blockIdx.x * blockDim.x + threadIdx.x;
  if (g >= BN) return;
  float* row = in7 + (long long)g * 32;
  row[0] = xyz[(long long)g * 3 + 0];
  row[1] = xyz[(long long)g * 3 + 1];
  row[2] = xyz[(long long)g * 3 + 2];
  row[3] = normals[(long long)g * 3 + 0];
  row[4] = normals[(long long)g * 3 + 1];
  row[5] = normals[(long long)g * 3 + 2];
  row[6] = curv[g];
#pragma unroll
  for (int j = 7; j < 32; ++j) row[j] = 0.0f;
}

// ---------------------------------------------------------------------
// Fused point-transformer attention layer. One wave == one query point,
// its 16 neighbors == the 16-row WMMA M-tile.
//   pos  = relu(rel @ d_w1 + d_b1) @ d_w2 + d_b2              (16 x 128)
//   a    = q_p - k_n + pos
//   attn = softmax_over_16( relu(a @ g_w1 + g_b1) @ g_w2 + g_b2 )
//   out  = sum_m attn * (v_n + pos)                           (1 x 128)
// All weights pre-packed to B-fragment layout (32B/lane contiguous loads).
// ---------------------------------------------------------------------
__global__ void attn_layer_kernel(const float* __restrict__ qmat,
                                  const float* __restrict__ kmat,
                                  const float* __restrict__ vmat,
                                  const float* __restrict__ xyz,
                                  const int* __restrict__ idx,
                                  const _Float16* __restrict__ dw1p,
                                  const float* __restrict__ db1,
                                  const _Float16* __restrict__ dw2p,
                                  const float* __restrict__ db2,
                                  const _Float16* __restrict__ gw1p,
                                  const float* __restrict__ gb1,
                                  const _Float16* __restrict__ gw2p,
                                  const float* __restrict__ gb2,
                                  float* __restrict__ attnout) {
  __shared__ _Float16 hidH[8][16 * CH];   // per-wave f16 staging (4KB each)
  const int lane = threadIdx.x & 31;
  const int wave = threadIdx.x >> 5;
  const int wpb = blockDim.x >> 5;
  _Float16* hid = &hidH[wave][0];
  const int khi = (lane >> 4) & 1;
  const int nlo = lane & 15;

  for (long long p = (long long)blockIdx.x * wpb + wave; p < BN;
       p += (long long)gridDim.x * wpb) {
    // neighbor ids this lane touches as D-fragment rows (m = r + 8*khi)
    int nidx[8];
#pragma unroll
    for (int r = 0; r < 8; ++r) nidx[r] = idx[p * NK + r + 8 * khi];
#pragma unroll
    for (int r = 0; r < 8; ++r)
      __builtin_prefetch(vmat + (long long)nidx[r] * CH, 0, 0);

    // ---- stage 1: A_rel fragment (16x3 zero-padded to 16x32)
    hv16 arel;
#pragma unroll
    for (int e = 0; e < 16; ++e) arel[e] = (_Float16)0.0f;
    if (khi == 0) {                    // lanes>=16 hold K>=8, all zero
      long long ng = idx[p * NK + nlo];
      arel[0] = (_Float16)(xyz[p * 3 + 0] - xyz[ng * 3 + 0]);
      arel[1] = (_Float16)(xyz[p * 3 + 1] - xyz[ng * 3 + 1]);
      arel[2] = (_Float16)(xyz[p * 3 + 2] - xyz[ng * 3 + 2]);
    }

    // hidden1 = relu(rel @ d_w1 + d_b1) -> LDS f16
#pragma unroll
    for (int nt = 0; nt < 8; ++nt) {
      fv8 acc = {};
      acc = wmma_f16(arel, load_b_packed(dw1p, 8, 0, nt, lane), acc);
      int n = (nt << 4) + nlo;
      float bv = db1[n];
#pragma unroll
      for (int r = 0; r < 8; ++r)
        hid[(r + 8 * khi) * CH + n] = (_Float16)fmaxf(acc[r] + bv, 0.0f);
    }

    // ---- stage 2: pos = hidden1 @ d_w2 + d_b2 (registers, D layout)
    hv16 Af[4];
#pragma unroll
    for (int kc = 0; kc < 4; ++kc) Af[kc] = load_a_lds(hid, kc * 32, lane);
    fv8 pos[8];
#pragma unroll
    for (int nt = 0; nt < 8; ++nt) {
      fv8 acc = {};
#pragma unroll
      for (int kc = 0; kc < 4; ++kc)
        acc = wmma_f16(Af[kc], load_b_packed(dw2p, 8, kc, nt, lane), acc);
      float bv = db2[(nt << 4) + nlo];
#pragma unroll
      for (int r = 0; r < 8; ++r) acc[r] += bv;
      pos[nt] = acc;
    }

    // ---- stage 3: a = q_p - k_n + pos -> LDS f16
    const float* qp = qmat + p * CH;
#pragma unroll
    for (int nt = 0; nt < 8; ++nt) {
      int n = (nt << 4) + nlo;
      float qv = qp[n];
#pragma unroll
      for (int r = 0; r < 8; ++r) {
        float kv = kmat[(long long)nidx[r] * CH + n];
        hid[(r + 8 * khi) * CH + n] = (_Float16)(qv - kv + pos[nt][r]);
      }
    }

    // ---- stage 4: h2 = relu(a @ g_w1 + g_b1) -> LDS f16 (A frags cached first)
#pragma unroll
    for (int kc = 0; kc < 4; ++kc) Af[kc] = load_a_lds(hid, kc * 32, lane);
#pragma unroll
    for (int nt = 0; nt < 8; ++nt) {
      fv8 acc = {};
#pragma unroll
      for (int kc = 0; kc < 4; ++kc)
        acc = wmma_f16(Af[kc], load_b_packed(gw1p, 8, kc, nt, lane), acc);
      int n = (nt << 4) + nlo;
      float bv = gb1[n];
#pragma unroll
      for (int r = 0; r < 8; ++r)
        hid[(r + 8 * khi) * CH + n] = (_Float16)fmaxf(acc[r] + bv, 0.0f);
    }

    // ---- stage 5: logits = h2 @ g_w2 + g_b2 (registers)
#pragma unroll
    for (int kc = 0; kc < 4; ++kc) Af[kc] = load_a_lds(hid, kc * 32, lane);
    fv8 att[8];
#pragma unroll
    for (int nt = 0; nt < 8; ++nt) {
      fv8 acc = {};
#pragma unroll
      for (int kc = 0; kc < 4; ++kc)
        acc = wmma_f16(Af[kc], load_b_packed(gw2p, 8, kc, nt, lane), acc);
      float bv = gb2[(nt << 4) + nlo];
#pragma unroll
      for (int r = 0; r < 8; ++r) acc[r] += bv;
      att[nt] = acc;
    }

    // ---- stage 6: softmax over the 16 neighbors per channel.
    // Column n lives in regs r=0..7 of lanes n and n+16 -> reduce + shfl_xor(16).
#pragma unroll
    for (int nt = 0; nt < 8; ++nt) {
      float mx = att[nt][0];
#pragma unroll
      for (int r = 1; r < 8; ++r) mx = fmaxf(mx, att[nt][r]);
      mx = fmaxf(mx, __shfl_xor(mx, 16, 32));
      float s = 0.0f;
#pragma unroll
      for (int r = 0; r < 8; ++r) {
        float e = __expf(att[nt][r] - mx);
        att[nt][r] = e;
        s += e;
      }
      s += __shfl_xor(s, 16, 32);
      float inv = 1.0f / s;
#pragma unroll
      for (int r = 0; r < 8; ++r) att[nt][r] *= inv;
    }

    // ---- stage 7: out[n] = sum_m attn * (v_n + pos)
#pragma unroll
    for (int nt = 0; nt < 8; ++nt) {
      int n = (nt << 4) + nlo;
      float s = 0.0f;
#pragma unroll
      for (int r = 0; r < 8; ++r) {
        float vv = vmat[(long long)nidx[r] * CH + n] + pos[nt][r];
        s += att[nt][r] * vv;
      }
      s += __shfl_xor(s, 16, 32);
      if (lane < 16) attnout[p * CH + n] = s;
    }
  }
}

// refined = normalize(normals + delta)
__global__ void finalize_normals_kernel(const float* __restrict__ normals,
                                        const float* __restrict__ delta,
                                        float* __restrict__ out) {
  int g = blockIdx.x * blockDim.x + threadIdx.x;
  if (g >= BN) return;
  float x = normals[(long long)g * 3 + 0] + delta[(long long)g * 3 + 0];
  float y = normals[(long long)g * 3 + 1] + delta[(long long)g * 3 + 1];
  float z = normals[(long long)g * 3 + 2] + delta[(long long)g * 3 + 2];
  float n = sqrtf(x * x + y * y + z * z);
  n = fmaxf(n, 1e-12f);
  float inv = 1.0f / n;
  out[(long long)g * 3 + 0] = x * inv;
  out[(long long)g * 3 + 1] = y * inv;
  out[(long long)g * 3 + 2] = z * inv;
}

// ---------------------------------------------------------------------
extern "C" void kernel_launch(void* const* d_in, const int* in_sizes, int n_in,
                              void* d_out, int out_size, void* d_ws, size_t ws_size,
                              hipStream_t stream) {
  (void)in_sizes; (void)n_in; (void)out_size; (void)ws_size;
  const float* xyz     = (const float*)d_in[0];
  const float* normals = (const float*)d_in[1];
  const float* emb_w1  = (const float*)d_in[2];
  const float* emb_b1  = (const float*)d_in[3];
  const float* emb_w2  = (const float*)d_in[4];
  const float* emb_b2  = (const float*)d_in[5];
  const float* q_w = (const float*)d_in[6],  *q_b = (const float*)d_in[7];
  const float* k_w = (const float*)d_in[8],  *k_b = (const float*)d_in[9];
  const float* v_w = (const float*)d_in[10], *v_b = (const float*)d_in[11];
  const float* d_w1 = (const float*)d_in[12], *d_b1 = (const float*)d_in[13];
  const float* d_w2 = (const float*)d_in[14], *d_b2 = (const float*)d_in[15];
  const float* g_w1 = (const float*)d_in[16], *g_b1 = (const float*)d_in[17];
  const float* g_w2 = (const float*)d_in[18], *g_b2 = (const float*)d_in[19];
  const float* o_w  = (const float*)d_in[20], *o_b  = (const float*)d_in[21];
  const float* nh_w1 = (const float*)d_in[22], *nh_b1 = (const float*)d_in[23];
  const float* nh_w2 = (const float*)d_in[24], *nh_b2 = (const float*)d_in[25];
  const float* nh_w3 = (const float*)d_in[26], *nh_b3 = (const float*)d_in[27];
  const float* ch_w1 = (const float*)d_in[28], *ch_b1 = (const float*)d_in[29];
  const float* ch_w2 = (const float*)d_in[30], *ch_b2 = (const float*)d_in[31];
  float* out = (float*)d_out;

  // workspace carve-out
  char* ws = (char*)d_ws;
  size_t off = 0;
  auto alloc = [&](size_t bytes) -> void* {
    void* p = ws + off;
    off = (off + bytes + 255) & ~(size_t)255;
    return p;
  };
  int*   idx     = (int*)  alloc((size_t)BN * NK * sizeof(int));
  float* curv    = (float*)alloc((size_t)BN * sizeof(float));
  float* in7     = (float*)alloc((size_t)BN * 32 * sizeof(float));
  float* feats   = (float*)alloc((size_t)BN * CH * sizeof(float));
  float* featsB  = (float*)alloc((size_t)BN * CH * sizeof(float));
  float* tbuf    = (float*)alloc((size_t)BN * CH * sizeof(float));
  float* qb      = (float*)alloc((size_t)BN * CH * sizeof(float));
  float* kb      = (float*)alloc((size_t)BN * CH * sizeof(float));
  float* vb      = (float*)alloc((size_t)BN * CH * sizeof(float));
  float* attnout = (float*)alloc((size_t)BN * CH * sizeof(float));
  float* h64     = (float*)alloc((size_t)BN * 64 * sizeof(float));
  float* delta   = (float*)alloc((size_t)BN * 3 * sizeof(float));

  // pack weights (f32 row-major -> f16 B-fragment layout, zero padded)
  auto packW = [&](const float* W, int K, int N) -> _Float16* {
    int kcN = (K + 31) >> 5;
    int tn = (N + 15) >> 4;
    int threads = kcN * tn * 32;
    _Float16* p = (_Float16*)alloc((size_t)threads * 16 * sizeof(_Float16));
    pack_w_kernel<<<(threads + 255) / 256, 256, 0, stream>>>(W, K, N, p);
    return p;
  };

  _Float16* emb1p = packW(emb_w1, 7, CH);
  _Float16* emb2p = packW(emb_w2, CH, CH);
  _Float16 *qwp[NL], *kwp[NL], *vwp[NL], *owp[NL];
  _Float16 *dw1p[NL], *dw2p[NL], *gw1p[NL], *gw2p[NL];
  for (int l = 0; l < NL; ++l) {
    qwp[l]  = packW(q_w  + (size_t)l * CH * CH, CH, CH);
    kwp[l]  = packW(k_w  + (size_t)l * CH * CH, CH, CH);
    vwp[l]  = packW(v_w  + (size_t)l * CH * CH, CH, CH);
    owp[l]  = packW(o_w  + (size_t)l * CH * CH, CH, CH);
    dw1p[l] = packW(d_w1 + (size_t)l * 3 * CH, 3, CH);
    dw2p[l] = packW(d_w2 + (size_t)l * CH * CH, CH, CH);
    gw1p[l] = packW(g_w1 + (size_t)l * CH * CH, CH, CH);
    gw2p[l] = packW(g_w2 + (size_t)l * CH * CH, CH, CH);
  }
  _Float16* nh1p = packW(nh_w1, CH, 128);
  _Float16* nh2p = packW(nh_w2, 128, 64);
  _Float16* nh3p = packW(nh_w3, 64, 3);
  _Float16* ch1p = packW(ch_w1, CH, 64);
  _Float16* ch2p = packW(ch_w2, 64, 1);

  auto lingrid = [](int M, int N) -> int {
    long long tiles = (long long)(M / 16) * ((N + 15) / 16);
    long long blocks = (tiles + 7) / 8;     // 8 waves / 256-thread block
    if (blocks > 4096) blocks = 4096;       // grid-stride covers the rest
    return (int)blocks;
  };

  // 1) KNN + curvature (shared across layers: xyz never changes)
  knn_curv_kernel<<<dim3(NPTS / 256, BATCH), 256, 0, stream>>>(xyz, normals, idx, curv);

  // 2) input embedding
  pack_in7_kernel<<<(BN + 255) / 256, 256, 0, stream>>>(xyz, normals, curv, in7);
  linear_wmma_kernel<<<lingrid(BN, CH), 256, 0, stream>>>(
      in7, 32, emb1p, emb_b1, nullptr, tbuf, BN, 1, CH, 1);
  linear_wmma_kernel<<<lingrid(BN, CH), 256, 0, stream>>>(
      tbuf, CH, emb2p, emb_b2, nullptr, feats, BN, 4, CH, 0);

  // 3) transformer layers
  float* cur = feats;
  float* nxt = featsB;
  for (int l = 0; l < NL; ++l) {
    linear_wmma_kernel<<<lingrid(BN, CH), 256, 0, stream>>>(
        cur, CH, qwp[l], q_b + (size_t)l * CH, nullptr, qb, BN, 4, CH, 0);
    linear_wmma_kernel<<<lingrid(BN, CH), 256, 0, stream>>>(
        cur, CH, kwp[l], k_b + (size_t)l * CH, nullptr, kb, BN, 4, CH, 0);
    linear_wmma_kernel<<<lingrid(BN, CH), 256, 0, stream>>>(
        cur, CH, vwp[l], v_b + (size_t)l * CH, nullptr, vb, BN, 4, CH, 0);

    attn_layer_kernel<<<2048, 256, 0, stream>>>(
        qb, kb, vb, xyz, idx,
        dw1p[l], d_b1 + (size_t)l * CH,
        dw2p[l], d_b2 + (size_t)l * CH,
        gw1p[l], g_b1 + (size_t)l * CH,
        gw2p[l], g_b2 + (size_t)l * CH,
        attnout);

    linear_wmma_kernel<<<lingrid(BN, CH), 256, 0, stream>>>(
        attnout, CH, owp[l], o_b + (size_t)l * CH,
        /*residual=*/cur, nxt, BN, 4, CH, 0);
    float* t = cur; cur = nxt; nxt = t;
  }

  // 4) normal head: relu(C->128) -> relu(128->64) -> 64->3, normalize
  linear_wmma_kernel<<<lingrid(BN, 128), 256, 0, stream>>>(
      cur, CH, nh1p, nh_b1, nullptr, tbuf, BN, 4, 128, 1);
  linear_wmma_kernel<<<lingrid(BN, 64), 256, 0, stream>>>(
      tbuf, 128, nh2p, nh_b2, nullptr, h64, BN, 4, 64, 1);
  linear_wmma_kernel<<<lingrid(BN, 3), 256, 0, stream>>>(
      h64, 64, nh3p, nh_b3, nullptr, delta, BN, 2, 3, 0);
  finalize_normals_kernel<<<(BN + 255) / 256, 256, 0, stream>>>(normals, delta, out);

  // 5) confidence head: relu(C->64) -> sigmoid(64->1), written after normals
  linear_wmma_kernel<<<lingrid(BN, 64), 256, 0, stream>>>(
      cur, CH, ch1p, ch_b1, nullptr, h64, BN, 4, 64, 1);
  linear_wmma_kernel<<<lingrid(BN, 1), 256, 0, stream>>>(
      h64, 64, ch2p, ch_b2, nullptr, out + (size_t)BN * 3, BN, 2, 1, 2);
}
